// LSTM_12137577578552
// MI455X (gfx1250) — matrix operand order repaired and checked by reading, not compiled, not run on hardware
//
#include <hip/hip_runtime.h>
#include <hip/hip_bf16.h>
#include <math.h>

typedef __attribute__((ext_vector_type(16))) __bf16 v16bf;
typedef __attribute__((ext_vector_type(8)))  float  v8f;

#define HIDDEN 128
#define GATES  512          // 4*HIDDEN, torch gate order i,f,g,o
#define NIN    6
#define TLEN   1024
#define BTOT   512
#define BTILE  16
#define NKK    5            // K padded to 160 = 5*32 (128 h + 6 x + 1 bias + pad)
#define NCT    32           // GATES/16 column tiles
#define NGATE  4

// ---- LDS layout (bytes) ----
#define WB_ELEMS (NCT*NKK*32*16)          // 81920 bf16 -> 163840 B (B fragments)
#define AH_ELEMS (NKK*32*16)              //  2560 bf16 ->   5120 B (one A buffer)
#define OFF_WB    0
#define OFF_AH    (WB_ELEMS*2)                     // double buffered: 2*5120 B
#define OFF_HFIN  (OFF_AH + 2*AH_ELEMS*2)          // 16*128 f32 = 8192 B
#define OFF_LOGIT (OFF_HFIN + BTILE*HIDDEN*4)      // 16*6 f32
#define LDS_BYTES (OFF_LOGIT + BTILE*NIN*4)        // ~178 KB (< 320 KB/WGP)

// ---- fast activations: CDNA5 hardware V_TANH_F32 when available ----
#if __has_builtin(__builtin_amdgcn_tanhf)
__device__ __forceinline__ float fast_tanh(float v) { return __builtin_amdgcn_tanhf(v); }
#elif __has_builtin(__builtin_amdgcn_tanh_f32)
__device__ __forceinline__ float fast_tanh(float v) { return __builtin_amdgcn_tanh_f32(v); }
#else
__device__ __forceinline__ float fast_tanh(float v) {
    float e = __expf(2.0f * v);
    return (e - 1.0f) / (e + 1.0f);
}
#endif
__device__ __forceinline__ float fast_sigmoid(float v) {
    return fmaf(0.5f, fast_tanh(0.5f * v), 0.5f);
}

__global__ void lstm_wmma_kernel(const float* __restrict__ x,
                                 const float* __restrict__ W_ih,
                                 const float* __restrict__ W_hh,
                                 const float* __restrict__ b_ih,
                                 const float* __restrict__ b_hh,
                                 const float* __restrict__ W_fc,
                                 const float* __restrict__ b_fc,
                                 float* __restrict__ out)
{
    extern __shared__ char smem[];
    __bf16* Wb     = (__bf16*)(smem + OFF_WB);    // [ct][kk][lane][16] B fragments
    __bf16* Ah     = (__bf16*)(smem + OFF_AH);    // [2][kk][lane][16]  A fragments (dbl buf)
    float*  hfin   = (float*)(smem + OFF_HFIN);   // [16][128]
    float*  logitsL= (float*)(smem + OFF_LOGIT);  // [16][6]

    const int tid  = threadIdx.x;
    const int lane = tid & 31;
    const int wave = tid >> 5;
    const int row0 = blockIdx.x * BTILE;

    // ---- Stage augmented B: B[k,n] = W_hh[n,k] | W_ih[n,k-128] | bias | 0 ----
    // B k order used here (contiguous, same as A): lanes 0-15 -> K = 32*kk + s ; lanes 16-31 -> K = 32*kk + 16 + s
    for (int idx = tid; idx < WB_ELEMS; idx += blockDim.x) {
        int s  = idx & 15;
        int ln = (idx >> 4) & 31;
        int kk = (idx >> 9) % NKK;
        int ct = idx / (NKK * 512);
        int K  = kk * 32 + ((ln >> 4) << 4) + s;
        int N  = ct * 16 + (ln & 15);
        float v;
        if      (K < HIDDEN)        v = W_hh[N * HIDDEN + K];
        else if (K < HIDDEN + NIN)  v = W_ih[N * NIN + (K - HIDDEN)];
        else if (K == HIDDEN + NIN) v = b_ih[N] + b_hh[N];
        else                        v = 0.0f;
        Wb[idx] = (__bf16)v;
    }

    // ---- Init both A buffers: h=0 (kk 0..3); kk 4: x_0 | 1.0 | zero pad ----
    for (int idx = tid; idx < AH_ELEMS; idx += blockDim.x) {
        int s  = idx & 15;
        int ln = (idx >> 4) & 31;
        int kk = idx >> 9;
        float v = 0.0f;
        if (kk == 4 && ln < 16) {
            if (s < NIN)       v = x[((size_t)(row0 + ln) * TLEN + 0) * NIN + s];
            else if (s == NIN) v = 1.0f;
        }
        __bf16 b = (__bf16)v;
        Ah[idx] = b;
        Ah[AH_ELEMS + idx] = b;   // pads/bias valid in both parities
    }
    __syncthreads();

    // This thread's fixed hidden-column and A-scatter coordinates.
    // wave w owns column tiles {w, 8+w, 16+w, 24+w} => i/f/g/o tiles for the
    // SAME 16 hidden columns -> LSTM cell update entirely in registers.
    const int colj  = (wave << 4) + (lane & 15);      // j in [0,128)
    const int kkj   = colj >> 5;
    const int remj  = colj & 31;
    const int lnOff = ((remj >> 4) & 1) ? 16 : 0;
    const int slotj = remj & 15;
    const int rbase = (lane >> 4) * 8;
    const bool xldr = (tid < BTILE * NIN);
    const int xr = tid / NIN, xi = tid % NIN;

    float c_reg[8];
#pragma unroll
    for (int q = 0; q < 8; ++q) c_reg[q] = 0.0f;

    for (int t = 0; t < TLEN; ++t) {
        const __bf16* AhR = Ah + (t & 1) * AH_ELEMS;
        __bf16*       AhW = Ah + ((t + 1) & 1) * AH_ELEMS;

        // Issue next-step x load early so global latency hides under the WMMAs
        float xv = 0.0f;
        if (t + 1 < TLEN && xldr)
            xv = x[((size_t)(row0 + xr) * TLEN + (t + 1)) * NIN + xi];

        // ---- WMMA phase: [h, x, 1] @ B_aug ----
        v16bf afrag[NKK];
#pragma unroll
        for (int kk = 0; kk < NKK; ++kk)
            afrag[kk] = *(const v16bf*)(AhR + (kk * 32 + lane) * 16);

        v8f acc[NGATE];
#pragma unroll
        for (int g = 0; g < NGATE; ++g) {
            const int ct = g * 8 + wave;
            acc[g] = (v8f){};
#pragma unroll
            for (int kk = 0; kk < NKK; ++kk) {
                v16bf bfrag = *(const v16bf*)(Wb + ((ct * NKK + kk) * 32 + lane) * 16);
                acc[g] = __builtin_amdgcn_wmma_f32_16x16x32_bf16(
                           false, afrag[kk], false, bfrag,
                           (short)0, acc[g], false, false);
            }
        }

        // ---- LSTM cell update in registers (i=acc0, f=acc1, g=acc2, o=acc3) ----
#pragma unroll
        for (int i = 0; i < 8; ++i) {
            float cn = fast_sigmoid(acc[1][i]) * c_reg[i]
                     + fast_sigmoid(acc[0][i]) * fast_tanh(acc[2][i]);
            c_reg[i] = cn;
            float h  = fast_sigmoid(acc[3][i]) * fast_tanh(cn);
            const int r = rbase + i;
            AhW[(kkj * 32 + r + lnOff) * 16 + slotj] = (__bf16)h;
            if (t == TLEN - 1) hfin[r * HIDDEN + colj] = h;
        }

        // ---- stage x_{t+1} into write buffer (bias slot & pads persist) ----
        if (t + 1 < TLEN && xldr)
            AhW[(4 * 32 + xr) * 16 + xi] = (__bf16)xv;

        __syncthreads();   // single barrier per step (A is double buffered)
    }

    // ---- final FC + softmax ----
    if (tid < BTILE * NIN) {
        const int r = tid / NIN, o = tid % NIN;
        float acc = b_fc[o];
        for (int k = 0; k < HIDDEN; ++k)
            acc += hfin[r * HIDDEN + k] * W_fc[o * HIDDEN + k];
        logitsL[r * NIN + o] = acc;
    }
    __syncthreads();
    if (tid < BTILE) {
        float m = -1e30f;
        for (int o = 0; o < NIN; ++o) m = fmaxf(m, logitsL[tid * NIN + o]);
        float ev[NIN], s = 0.0f;
        for (int o = 0; o < NIN; ++o) { ev[o] = __expf(logitsL[tid * NIN + o] - m); s += ev[o]; }
        float inv = 1.0f / s;
        for (int o = 0; o < NIN; ++o)
            out[(size_t)(row0 + tid) * NIN + o] = ev[o] * inv;
    }
}

extern "C" void kernel_launch(void* const* d_in, const int* in_sizes, int n_in,
                              void* d_out, int out_size, void* d_ws, size_t ws_size,
                              hipStream_t stream) {
    (void)in_sizes; (void)n_in; (void)out_size; (void)d_ws; (void)ws_size;
    const float* x    = (const float*)d_in[0];
    const float* W_ih = (const float*)d_in[1];
    const float* W_hh = (const float*)d_in[2];
    const float* b_ih = (const float*)d_in[3];
    const float* b_hh = (const float*)d_in[4];
    const float* W_fc = (const float*)d_in[5];
    const float* b_fc = (const float*)d_in[6];
    float* out = (float*)d_out;

    dim3 grid(BTOT / BTILE);   // 32 workgroups, one WGP each (weights LDS resident)
    dim3 block(256);           // 8 waves of 32
    lstm_wmma_kernel<<<grid, block, LDS_BYTES, stream>>>(
        x, W_ih, W_hh, b_ih, b_hh, W_fc, b_fc, out);
}
